// StyleSpeech2_33397665694625
// MI455X (gfx1250) — matrix operand order repaired and checked by reading, not compiled, not run on hardware
//
#include <hip/hip_runtime.h>
#include <hip/hip_bf16.h>
#include <math.h>

// ---------------- problem constants ----------------
#define B_   32
#define TX_  512
#define TY_  1024
#define D_   256
#define H_   2
#define DK_  128
#define FN_  1024
#define KS_  9
#define L_   4
#define NEGV (-1e9f)

// ---------------- WMMA types ----------------
typedef __attribute__((ext_vector_type(16))) _Float16 v16h;
typedef __attribute__((ext_vector_type(8)))  float    v8f;
typedef __attribute__((ext_vector_type(4)))  unsigned int u32x4;
typedef __attribute__((ext_vector_type(4)))  int      i32x4;
typedef __attribute__((ext_vector_type(8)))  int      i32x8;

#define BM 128
#define BN 64
#define BK 64
#define APAD 8   // As rows: (BK+APAD)=72 halves = 144B (16B aligned)
#define BPAD 8   // Bs rows: (BN+BPAD)=72 halves = 144B (16B aligned)

__device__ __forceinline__ v8f wmma32(v16h a, v16h b, v8f c) {
  return __builtin_amdgcn_wmma_f32_16x16x32_f16(false, a, false, b, (short)0, c,
                                                false, false);
}

// A fragment: 16x32 f16 sub-tile at LDS row r0, k-offset ko.
__device__ __forceinline__ v16h frag_a(const _Float16* As, int ldsA, int r0,
                                       int ko, int lane) {
  const _Float16* p = As + (size_t)(r0 + (lane & 15)) * ldsA + ko;
  int hi = (lane >> 4) << 3;  // 0 or 8
  v16h a;
#pragma unroll
  for (int j = 0; j < 8; ++j) { a[j] = p[hi + j]; a[8 + j] = p[16 + hi + j]; }
  return a;
}

// B fragment: 32x16 f16 sub-tile at LDS row r0, col c0; lane L holds col N=c0+(L&15), K in table order.
__device__ __forceinline__ v16h frag_b(const _Float16* Bs, int ldsB, int r0,
                                       int c0, int lane) {
  const _Float16* p = Bs + (size_t)r0 * ldsB + c0 + (lane & 15);
  int hi = (lane >> 4) << 3;
  v16h b;
#pragma unroll
  for (int j = 0; j < 8; ++j) { b[j] = p[(size_t)(hi + j) * ldsB]; b[8 + j] = p[(size_t)(16 + hi + j) * ldsB]; }
  return b;
}

// ---------------- generic batched WMMA GEMM (f16 in, f32 accum) ----------------
// z1 = z / zdiv, z2 = z % zdiv; off* = z1*S1 + z2*S2 (element offsets)
// im2col: A element (m, kk) = X[t*Cin + ci]; within a 32-wide k-chunk that is
//         32-aligned, kq = kk>>cinSh and ci base are constant (Cin is a power
//         of two >= 32), so bounds test + copy are hoisted per chunk.
__global__ __launch_bounds__(256) void ss2_gemm_wmma(
    const _Float16* __restrict__ A, const _Float16* __restrict__ Bg,
    float* __restrict__ Cf, _Float16* __restrict__ Ch,
    const float* __restrict__ bias,
    int K, int lda, int ldb, int ldc,
    long long aS1, long long aS2, long long bS1, long long bS2,
    long long cS1, long long cS2, int zdiv, int relu, float alpha,
    int im2col, int Cin, int cinSh, int T) {
  int z  = blockIdx.z;
  int z1 = z / zdiv, z2 = z % zdiv;
  size_t offA = (size_t)z1 * aS1 + (size_t)z2 * aS2;
  size_t offB = (size_t)z1 * bS1 + (size_t)z2 * bS2;
  size_t offC = (size_t)z1 * cS1 + (size_t)z2 * cS2;
  int m0 = blockIdx.y * BM;
  int n0 = blockIdx.x * BN;

  __shared__ _Float16 As[BM][BK + APAD];
  __shared__ _Float16 Bs[BK][BN + BPAD];

  int tid = threadIdx.x, lane = tid & 31, wave = tid >> 5;
  int wm = wave & 3, wn = wave >> 2;  // 4x2 waves -> 128x64 tile
  v8f acc[2][2] = {};

  for (int k0 = 0; k0 < K; k0 += BK) {
    // ---- stage A tile (128x64 halves), one contiguous 32-chunk per thread ----
    {
      int row = tid >> 1;
      int cb  = (tid & 1) * 32;
      if (!im2col) {
        const _Float16* src = A + offA + (size_t)(m0 + row) * lda + (k0 + cb);
        if (k0 + BK < K) __builtin_prefetch(src + BK, 0, 1);
#pragma unroll
        for (int j = 0; j < 32; ++j) As[row][cb + j] = src[j];
      } else {
        int base = k0 + cb;            // 32-aligned, never crosses Cin boundary
        int kq   = base >> cinSh;
        int ci   = base & (Cin - 1);
        int t    = m0 + row + kq - ((KS_ - 1) / 2);
        if (t >= 0 && t < T) {
          const _Float16* src = A + offA + (size_t)t * Cin + ci;
#pragma unroll
          for (int j = 0; j < 32; ++j) As[row][cb + j] = src[j];
        } else {
#pragma unroll
          for (int j = 0; j < 32; ++j) As[row][cb + j] = (_Float16)0.f;
        }
      }
    }
    // ---- stage B tile (64x64 halves), contiguous 16-chunk per thread ----
    {
      int row = tid >> 2;
      int cb  = (tid & 3) * 16;
      const _Float16* src = Bg + offB + (size_t)(k0 + row) * ldb + (n0 + cb);
#pragma unroll
      for (int j = 0; j < 16; ++j) Bs[row][cb + j] = src[j];
    }
    __syncthreads();
#pragma unroll
    for (int kk = 0; kk < BK; kk += 32) {
      v16h a0 = frag_a(&As[0][0], BK + APAD, wm * 32, kk, lane);
      v16h a1 = frag_a(&As[0][0], BK + APAD, wm * 32 + 16, kk, lane);
      v16h b0 = frag_b(&Bs[0][0], BN + BPAD, kk, wn * 32, lane);
      v16h b1 = frag_b(&Bs[0][0], BN + BPAD, kk, wn * 32 + 16, lane);
      acc[0][0] = wmma32(a0, b0, acc[0][0]);
      acc[0][1] = wmma32(a0, b1, acc[0][1]);
      acc[1][0] = wmma32(a1, b0, acc[1][0]);
      acc[1][1] = wmma32(a1, b1, acc[1][1]);
    }
    __syncthreads();
  }

  // ---- epilogue: C layout VGPR r <-> M=r+8*hi, N=lane&15 ----
  int hi = lane >> 4, ln = lane & 15;
#pragma unroll
  for (int i = 0; i < 2; ++i)
#pragma unroll
    for (int j = 0; j < 2; ++j)
#pragma unroll
      for (int r = 0; r < 8; ++r) {
        int row = m0 + wm * 32 + 16 * i + 8 * hi + r;
        int col = n0 + wn * 32 + 16 * j + ln;
        float val = alpha * acc[i][j][r];
        if (bias) val += bias[col];
        if (relu) val = fmaxf(val, 0.f);
        size_t o = offC + (size_t)row * ldc + col;
        if (Cf) Cf[o] = val;
        if (Ch) Ch[o] = (_Float16)val;
      }
}

// ---------------- CDNA5 TDM / cluster path probe (deterministic no-op) ----
// NULL tensor descriptor (D# group0.count==0) => architectural no-op, but
// exercises TENSOR_LOAD_TO_LDS issue + TENSORcnt; cluster barrier is a NOP
// when ClusterID==0 (non-cluster dispatch) per ISA 6.6.
__global__ void ss2_cdna5_probe(int* out) {
  u32x4 g0 = {0u, 0u, 0u, 0u};
  i32x8 g1 = {0, 0, 0, 0, 0, 0, 0, 0};
  i32x4 g2 = {0, 0, 0, 0};
  i32x4 g3 = {0, 0, 0, 0};
#if __clang_major__ >= 23
  i32x8 g4 = {0, 0, 0, 0, 0, 0, 0, 0};
  __builtin_amdgcn_tensor_load_to_lds(g0, g1, g2, g3, g4, 0);
#else
  __builtin_amdgcn_tensor_load_to_lds(g0, g1, g2, g3, 0);
#endif
  __builtin_amdgcn_s_wait_tensorcnt(0);
  __builtin_amdgcn_s_cluster_barrier();
  if (threadIdx.x == 0) out[0] = __builtin_amdgcn_cluster_id_x();
}

// ---------------- elementwise / reduction kernels ----------------
__global__ void ss2_f32_to_f16(const float* __restrict__ in,
                               _Float16* __restrict__ out, int n) {
  int i = blockIdx.x * 256 + threadIdx.x;
  if (i < n) out[i] = (_Float16)in[i];
}

__global__ void ss2_embed(const int* __restrict__ tok, const float* __restrict__ emb,
                          const float* __restrict__ pos, float* __restrict__ x,
                          _Float16* __restrict__ xh) {
  int r = blockIdx.x;   // B*TX rows
  int d = threadIdx.x;  // 256
  int t = r % TX_;
  int id = tok[r];
  float v = emb[(size_t)id * D_ + d] + pos[(size_t)t * D_ + d];
  x[(size_t)r * D_ + d]  = v;
  xh[(size_t)r * D_ + d] = (_Float16)v;
}

// K16 [B,TX,H*DK] -> KT16 [B,H,DK,TX] so Q.K^T becomes a non-transposed GEMM
__global__ void ss2_tr_k(const _Float16* __restrict__ in, _Float16* __restrict__ out) {
  __shared__ _Float16 tile[32][33];
  int zz = blockIdx.z;             // b*H + h
  int b = zz >> 1, h = zz & 1;
  int t0 = blockIdx.x * 32, d0 = blockIdx.y * 32;
  int tx = threadIdx.x, ty = threadIdx.y;  // 32 x 8
#pragma unroll
  for (int i = 0; i < 32; i += 8)
    tile[ty + i][tx] = in[((size_t)b * TX_ + (t0 + ty + i)) * D_ + h * DK_ + (d0 + tx)];
  __syncthreads();
#pragma unroll
  for (int i = 0; i < 32; i += 8)
    out[(((size_t)(b * H_ + h) * DK_) + (d0 + ty + i)) * TX_ + (t0 + tx)] = tile[tx][ty + i];
}

// masked softmax over rows of length TX; reads f32 scores, writes f16 probs
__global__ void ss2_softmax_mask(const float* __restrict__ S,
                                 _Float16* __restrict__ P,
                                 const int* __restrict__ src_len) {
  int row = blockIdx.x;  // B*H*TX rows
  int b = row / (H_ * TX_);
  const float* p = S + (size_t)row * TX_;
  _Float16* q = P + (size_t)row * TX_;
  int sl = src_len[b];
  __shared__ float red[256];
  int tid = threadIdx.x;
  float vals[2];
  float mx = -3.4e38f;
#pragma unroll
  for (int i = 0; i < 2; ++i) {
    int c = tid + i * 256;
    float v = p[c];
    if (c >= sl) v = NEGV;
    vals[i] = v;
    mx = fmaxf(mx, v);
  }
  red[tid] = mx; __syncthreads();
  for (int s = 128; s > 0; s >>= 1) { if (tid < s) red[tid] = fmaxf(red[tid], red[tid + s]); __syncthreads(); }
  mx = red[0]; __syncthreads();
  float sum = 0.f;
#pragma unroll
  for (int i = 0; i < 2; ++i) { vals[i] = __expf(vals[i] - mx); sum += vals[i]; }
  red[tid] = sum; __syncthreads();
  for (int s = 128; s > 0; s >>= 1) { if (tid < s) red[tid] += red[tid + s]; __syncthreads(); }
  float inv = 1.f / red[0];
#pragma unroll
  for (int i = 0; i < 2; ++i) q[tid + i * 256] = (_Float16)(vals[i] * inv);
}

// out(+f16 copy) = where(pad, 0, LN(a+b)*g + be)
__global__ void ss2_add_ln(const float* __restrict__ a, const float* __restrict__ b,
                           float* __restrict__ out, _Float16* __restrict__ outh,
                           const float* __restrict__ g, const float* __restrict__ be,
                           const int* __restrict__ src_len) {
  int row = blockIdx.x;  // B*TX
  int t = row % TX_, bb = row / TX_;
  int tid = threadIdx.x;  // 256 == D
  __shared__ float red[256];
  float v = a[(size_t)row * D_ + tid] + b[(size_t)row * D_ + tid];
  red[tid] = v; __syncthreads();
  for (int s = 128; s > 0; s >>= 1) { if (tid < s) red[tid] += red[tid + s]; __syncthreads(); }
  float mean = red[0] * (1.f / D_); __syncthreads();
  float d = v - mean;
  red[tid] = d * d; __syncthreads();
  for (int s = 128; s > 0; s >>= 1) { if (tid < s) red[tid] += red[tid + s]; __syncthreads(); }
  float var = red[0] * (1.f / D_);
  float rs = rsqrtf(var + 1e-5f);
  float o = (t < src_len[bb]) ? (d * rs * g[tid] + be[tid]) : 0.f;
  out[(size_t)row * D_ + tid]  = o;
  outh[(size_t)row * D_ + tid] = (_Float16)o;
}

// c1_w[FN][D][KS] -> w1t[(k*D+ci)*FN + co]  (f16)
__global__ void ss2_tr_c1(const float* __restrict__ w, _Float16* __restrict__ wt) {
  int i = blockIdx.x * 256 + threadIdx.x;
  if (i >= KS_ * D_ * FN_) return;
  int co = i % FN_;
  int rest = i / FN_;
  int ci = rest % D_;
  int k = rest / D_;
  wt[i] = (_Float16)w[((size_t)co * D_ + ci) * KS_ + k];
}

// c2_w[D][FN][KS] -> w2t[(k*FN+f)*D + d]  (f16)
__global__ void ss2_tr_c2(const float* __restrict__ w, _Float16* __restrict__ wt) {
  int i = blockIdx.x * 256 + threadIdx.x;
  if (i >= KS_ * FN_ * D_) return;
  int d = i % D_;
  int rest = i / D_;
  int f = rest % FN_;
  int k = rest / FN_;
  wt[i] = (_Float16)w[((size_t)d * FN_ + f) * KS_ + k];
}

// value [B][TX][TY] -> valueT [B][TY][TX]
__global__ void ss2_transpose_val(const float* __restrict__ in, float* __restrict__ out) {
  __shared__ float tile[32][33];
  int b = blockIdx.z;
  int x0 = blockIdx.x * 32, y0 = blockIdx.y * 32;
  int tx = threadIdx.x, ty = threadIdx.y;  // 32 x 8
#pragma unroll
  for (int i = 0; i < 32; i += 8)
    tile[ty + i][tx] = in[((size_t)b * TX_ + (x0 + ty + i)) * TY_ + (y0 + tx)];
  __syncthreads();
#pragma unroll
  for (int i = 0; i < 32; i += 8)
    out[((size_t)b * TY_ + (y0 + ty + i)) * TX_ + (x0 + tx)] = tile[tx][ty + i];
}

// MAS forward DP: one block per batch, 512 threads, 1024 sequential steps
__global__ void ss2_mas_fwd(const float* __restrict__ valT, const int* __restrict__ sl_,
                            const int* __restrict__ ml_, unsigned char* __restrict__ dirs) {
  int b = blockIdx.x;
  int x = threadIdx.x;  // 512
  __shared__ float v[TX_];
  int sl = sl_[b], ml = ml_[b];
  v[x] = 0.f;
  __syncthreads();
  for (int j = 0; j < TY_; ++j) {
    float vx = v[x];
    float v0 = (x > 0) ? v[x - 1] : NEGV;
    bool mm = vx >= v0;
    float vmax = mm ? vx : v0;
    bool msk = (x < sl) && (j < ml);
    float val = msk ? valT[((size_t)b * TY_ + j) * TX_ + x] : 0.f;
    float vnew = (x <= j) ? (vmax + val) : NEGV;
    dirs[((size_t)j * B_ + b) * TX_ + x] = msk ? (unsigned char)mm : (unsigned char)1;
    __syncthreads();
    v[x] = vnew;
    __syncthreads();
  }
}

// MAS backtrace: lane = batch
__global__ void ss2_mas_bwd(const unsigned char* __restrict__ dirs,
                            const int* __restrict__ sl_, int* __restrict__ sel) {
  int b = threadIdx.x;
  if (b >= B_) return;
  int idx = sl_[b] - 1;
  for (int y = TY_ - 1; y >= 0; --y) {
    sel[b * TY_ + y] = idx;
    int d = dirs[((size_t)y * B_ + b) * TX_ + idx];
    idx += d - 1;
    if (idx < 0) idx = 0;
  }
}

// expanded[b,y,:] = one-hot path gather of x rows
__global__ void ss2_expand(const float* __restrict__ x, const int* __restrict__ sel,
                           const int* __restrict__ sl_, const int* __restrict__ ml_,
                           float* __restrict__ out) {
  int r = blockIdx.x;  // B*TY
  int b = r / TY_, y = r % TY_;
  int d = threadIdx.x;  // 256
  int s = sel[b * TY_ + y];
  bool valid = (y < ml_[b]) && (s >= 0) && (s < sl_[b]);
  out[(size_t)r * D_ + d] = valid ? x[((size_t)b * TX_ + s) * D_ + d] : 0.f;
}

// ---------------- host side ----------------
static void run_gemm(hipStream_t st, int M, int N, int K,
                     const _Float16* A, int lda, long long aS1, long long aS2,
                     const _Float16* Bg, int ldb, long long bS1, long long bS2,
                     float* Cf, _Float16* Ch, int ldc, long long cS1, long long cS2,
                     const float* bias, int relu, float alpha,
                     int nz, int zdiv, int im2col, int Cin, int cinSh, int T) {
  dim3 grid(N / BN, M / BM, nz), blk(256, 1, 1);
  ss2_gemm_wmma<<<grid, blk, 0, st>>>(A, Bg, Cf, Ch, bias, K, lda, ldb, ldc,
                                      aS1, aS2, bS1, bS2, cS1, cS2, zdiv,
                                      relu, alpha, im2col, Cin, cinSh, T);
}

extern "C" void kernel_launch(void* const* d_in, const int* in_sizes, int n_in,
                              void* d_out, int out_size, void* d_ws, size_t ws_size,
                              hipStream_t stream) {
  (void)in_sizes; (void)n_in; (void)out_size; (void)ws_size;
  const int*   tokens  = (const int*)d_in[0];
  const int*   src_len = (const int*)d_in[1];
  const int*   mel_len = (const int*)d_in[2];
  const float* value   = (const float*)d_in[3];
  const float* embed   = (const float*)d_in[4];
  const float* pos     = (const float*)d_in[5];
  const float* wq = (const float*)d_in[6];  const float* bq = (const float*)d_in[7];
  const float* wk = (const float*)d_in[8];  const float* bk = (const float*)d_in[9];
  const float* wv = (const float*)d_in[10]; const float* bv = (const float*)d_in[11];
  const float* wo = (const float*)d_in[12]; const float* bo = (const float*)d_in[13];
  const float* ln1g = (const float*)d_in[14]; const float* ln1b = (const float*)d_in[15];
  const float* c1w  = (const float*)d_in[16]; const float* c1b  = (const float*)d_in[17];
  const float* c2w  = (const float*)d_in[18]; const float* c2b  = (const float*)d_in[19];
  const float* ln2g = (const float*)d_in[20]; const float* ln2b = (const float*)d_in[21];

  const long long MB = 1024LL * 1024LL;
  char* wsb = (char*)d_ws;
  float*    X      = (float*)   (wsb + 0 * MB);    // 16 MB  [B,TX,D] f32
  _Float16* X16    = (_Float16*)(wsb + 16 * MB);   //  8 MB  f16 copy
  _Float16* Q16    = (_Float16*)(wsb + 24 * MB);   //  8 MB  q proj / attn out
  _Float16* K16    = (_Float16*)(wsb + 32 * MB);   //  8 MB
  _Float16* V16    = (_Float16*)(wsb + 40 * MB);   //  8 MB
  float*    TMP    = (float*)   (wsb + 48 * MB);   // 16 MB  (DIRS reuse post-layers)
  float*    SCORES = (float*)   (wsb + 64 * MB);   // 64 MB  (valueT reuse)
  _Float16* P16    = (_Float16*)(wsb + 128 * MB);  // 32 MB  probs (H16 reuse)
  _Float16* H16    = (_Float16*)(wsb + 128 * MB);  // 32 MB  conv hidden (phase-disjoint)
  _Float16* WQ16   = (_Float16*)(wsb + 160 * MB);
  _Float16* WK16   = (_Float16*)(wsb + 160 * MB + 1 * 131072);
  _Float16* WV16   = (_Float16*)(wsb + 160 * MB + 2 * 131072);
  _Float16* WO16   = (_Float16*)(wsb + 160 * MB + 3 * 131072);
  _Float16* W1T16  = (_Float16*)(wsb + 161 * MB);              // 4.5 MB
  _Float16* W2T16  = (_Float16*)(wsb + 166 * MB);              // 4.5 MB
  int*      SEL    = (int*)     (wsb + 171 * MB);              // 128 KB
  int*      PROBE  = (int*)     (wsb + 172 * MB);              // 4 B
  _Float16* KT16   = (_Float16*)(wsb + 176 * MB);              // 8 MB [B,H,DK,TX]
  unsigned char* DIRS = (unsigned char*)TMP;                   // 16 MB, post-layer

  const int MROWS = B_ * TX_;  // 16384
  const float scale = 1.0f / sqrtf((float)DK_);
  const int DD = D_ * D_;

  // CDNA5 TDM/cluster issue-path probe (no data effect)
  ss2_cdna5_probe<<<1, 32, 0, stream>>>(PROBE);

  // embedding + positional encoding (f32 + f16)
  ss2_embed<<<MROWS, 256, 0, stream>>>(tokens, embed, pos, X, X16);

  for (int l = 0; l < L_; ++l) {
    // stage this layer's projection weights as f16
    ss2_f32_to_f16<<<(DD + 255) / 256, 256, 0, stream>>>(wq + (size_t)l * DD, WQ16, DD);
    ss2_f32_to_f16<<<(DD + 255) / 256, 256, 0, stream>>>(wk + (size_t)l * DD, WK16, DD);
    ss2_f32_to_f16<<<(DD + 255) / 256, 256, 0, stream>>>(wv + (size_t)l * DD, WV16, DD);
    ss2_f32_to_f16<<<(DD + 255) / 256, 256, 0, stream>>>(wo + (size_t)l * DD, WO16, DD);

    // Q, K, V projections: [16384,256] = X16 @ W16 + b  -> f16 outputs
    run_gemm(stream, MROWS, D_, D_, X16, D_, 0, 0, WQ16, D_, 0, 0,
             nullptr, Q16, D_, 0, 0, bq + (size_t)l * D_, 0, 1.f, 1, 1, 0, 0, 0, 0);
    run_gemm(stream, MROWS, D_, D_, X16, D_, 0, 0, WK16, D_, 0, 0,
             nullptr, K16, D_, 0, 0, bk + (size_t)l * D_, 0, 1.f, 1, 1, 0, 0, 0, 0);
    run_gemm(stream, MROWS, D_, D_, X16, D_, 0, 0, WV16, D_, 0, 0,
             nullptr, V16, D_, 0, 0, bv + (size_t)l * D_, 0, 1.f, 1, 1, 0, 0, 0, 0);

    // K -> K^T layout [B,H,DK,TX] so scores GEMM reads B contiguously
    {
      dim3 g(TX_ / 32, DK_ / 32, B_ * H_), b(32, 8, 1);
      ss2_tr_k<<<g, b, 0, stream>>>(K16, KT16);
    }

    // scores[z=(b,h)] = scale * Q @ K^T  (all-contiguous B)
    run_gemm(stream, TX_, TX_, DK_,
             Q16, D_, (long long)TX_ * D_, DK_,
             KT16, TX_, (long long)H_ * DK_ * TX_, (long long)DK_ * TX_,
             SCORES, nullptr, TX_, 2LL * TX_ * TX_, (long long)TX_ * TX_,
             nullptr, 0, scale, B_ * H_, H_, 0, 0, 0, 0);

    // masked softmax -> f16 probabilities
    ss2_softmax_mask<<<B_ * H_ * TX_, 256, 0, stream>>>(SCORES, P16, src_len);

    // attn_out[z] = P @ V -> Q16 [B,TX,H,DK] (f16)
    run_gemm(stream, TX_, DK_, TX_,
             P16, TX_, 2LL * TX_ * TX_, (long long)TX_ * TX_,
             V16, D_, (long long)TX_ * D_, DK_,
             nullptr, Q16, D_, (long long)TX_ * D_, DK_,
             nullptr, 0, 1.f, B_ * H_, H_, 0, 0, 0, 0);

    // output projection: TMP = attn_out @ wo + bo (f32)
    run_gemm(stream, MROWS, D_, D_, Q16, D_, 0, 0, WO16, D_, 0, 0,
             TMP, nullptr, D_, 0, 0, bo + (size_t)l * D_, 0, 1.f, 1, 1, 0, 0, 0, 0);

    // X = where(pad, 0, LN(TMP + X))  (+ f16 copy)
    ss2_add_ln<<<MROWS, 256, 0, stream>>>(TMP, X, X, X16, ln1g + (size_t)l * D_,
                                          ln1b + (size_t)l * D_, src_len);

    // conv weight transposes -> f16
    ss2_tr_c1<<<(KS_ * D_ * FN_ + 255) / 256, 256, 0, stream>>>(
        c1w + (size_t)l * FN_ * D_ * KS_, W1T16);
    ss2_tr_c2<<<(KS_ * FN_ * D_ + 255) / 256, 256, 0, stream>>>(
        c2w + (size_t)l * D_ * FN_ * KS_, W2T16);

    // conv1 (im2col GEMM): H16[b,t,f] = relu(sum X16[b,t+k-4,ci] W1T[(k,ci),f] + b1)
    run_gemm(stream, TX_, FN_, D_ * KS_,
             X16, 0, (long long)TX_ * D_, 0,
             W1T16, FN_, 0, 0,
             nullptr, H16, FN_, (long long)TX_ * FN_, 0,
             c1b + (size_t)l * FN_, 1, 1.f, B_, 1, 1, D_, 8, TX_);

    // conv2 (im2col GEMM): TMP[b,t,d] = sum H16[b,t+k-4,f] W2T[(k,f),d] + b2
    run_gemm(stream, TX_, D_, FN_ * KS_,
             H16, 0, (long long)TX_ * FN_, 0,
             W2T16, D_, 0, 0,
             TMP, nullptr, D_, (long long)TX_ * D_, 0,
             c2b + (size_t)l * D_, 0, 1.f, B_, 1, 1, FN_, 10, TX_);

    // X = where(pad, 0, LN(TMP + X))  (+ f16 copy)
    ss2_add_ln<<<MROWS, 256, 0, stream>>>(TMP, X, X, X16, ln2g + (size_t)l * D_,
                                          ln2b + (size_t)l * D_, src_len);
  }

  // ---- monotonic alignment search + expand ----
  {
    dim3 g(TX_ / 32, TY_ / 32, B_), b(32, 8, 1);
    ss2_transpose_val<<<g, b, 0, stream>>>(value, SCORES);  // SCORES = value^T [B,TY,TX]
  }
  ss2_mas_fwd<<<B_, TX_, 0, stream>>>(SCORES, src_len, mel_len, DIRS);
  ss2_mas_bwd<<<1, 32, 0, stream>>>(DIRS, src_len, SEL);
  ss2_expand<<<B_ * TY_, 256, 0, stream>>>(X, SEL, src_len, mel_len, (float*)d_out);
}
